// CustomLinear_87978110091557
// MI455X (gfx1250) — matrix operand order repaired and checked by reading, not compiled, not run on hardware
//
#include <hip/hip_runtime.h>
#include <stdint.h>

typedef __attribute__((ext_vector_type(16))) _Float16 v16h;
typedef __attribute__((ext_vector_type(8)))  _Float16 v8h;
typedef __attribute__((ext_vector_type(4)))  _Float16 v4h;
typedef __attribute__((ext_vector_type(8)))  float    v8f;
typedef __attribute__((ext_vector_type(4)))  float    v4f;

// Problem dims (fixed by the reference): M = B*S = 8192, K = 4096, N = 4096
constexpr int M_ = 8192, K_ = 4096, N_ = 4096;
constexpr int BM = 256, BN = 128, BK = 32;
constexpr int NKSTEP = K_ / BK;            // 128
constexpr int LDA = BK + 8;                // 40 f16 = 80B row stride: 16B-aligned, bank-rotating
constexpr int LDW = BK + 8;                // 40
constexpr int SA_ELEMS = 2 * BM * LDA;     // double-buffered A tile
constexpr int SW_ELEMS = 2 * BN * LDW;     // double-buffered W tile
constexpr size_t SMEM_BYTES = (size_t)(SA_ELEMS + SW_ELEMS + BN * 16) * sizeof(_Float16); // 64KB

__device__ __forceinline__ v16h make_v16(v8h lo, v8h hi) {
  return __builtin_shufflevector(lo, hi, 0,1,2,3,4,5,6,7,8,9,10,11,12,13,14,15);
}

__global__ __launch_bounds__(256)
void sqllm_gemm_f16wmma(const float* __restrict__ A, const int* __restrict__ QW,
                        const float* __restrict__ LUT, float* __restrict__ C) {
  extern __shared__ _Float16 smem[];
  _Float16* sA   = smem;                 // [2][BM][LDA]
  _Float16* sW   = sA + SA_ELEMS;        // [2][BN][LDW]
  _Float16* sLut = sW + SW_ELEMS;        // [BN][16]

  const int t    = threadIdx.x;
  const int lane = t & 31;
  const int wid  = t >> 5;               // 0..7
  const int wm   = (wid >> 1) * 64;      // wave M offset: 0,64,128,192
  const int wn   = (wid & 1) * 64;       // wave N offset: 0,64

  const int n0 = blockIdx.x * BN;
  const int m0 = blockIdx.y * BM;

  // ---- Stage per-block LUT rows into LDS as f16 ----
  {
    const int row = t >> 1;              // 0..127
    const int e0  = (t & 1) * 8;
    const float* src = LUT + (size_t)(n0 + row) * 16 + e0;
    #pragma unroll
    for (int i = 0; i < 8; ++i) sLut[row * 16 + e0 + i] = (_Float16)src[i];
  }

  // ---- staging registers for software pipeline ----
  v4f aReg[8];
  uint32_t qReg[2];
  const int arow = t >> 3;               // 0..31  (A: 8 lanes per row segment)
  const int acol = (t & 7) * 4;          // 0..28
  const int nn   = t >> 1;               // 0..127 (W: 2 int32s per thread)
  const int j2   = (t & 1) * 2;          // packed-dword pair index (0 or 2)

  auto load_stage = [&](int ks) {
    const float* ap = A + (size_t)(m0 + arow) * K_ + ks * BK + acol;
    #pragma unroll
    for (int r = 0; r < 8; ++r)
      aReg[r] = *(const v4f*)(ap + (size_t)r * 32 * K_);
    const int kp0 = ks * 4;              // qweight row = k/8
    qReg[0] = ((const uint32_t*)QW)[(size_t)(kp0 + j2    ) * N_ + n0 + nn];
    qReg[1] = ((const uint32_t*)QW)[(size_t)(kp0 + j2 + 1) * N_ + n0 + nn];
  };

  auto write_stage = [&](int buf) {
    _Float16* a = sA + buf * BM * LDA;
    #pragma unroll
    for (int r = 0; r < 8; ++r) {
      v4h h;
      #pragma unroll
      for (int i = 0; i < 4; ++i) h[i] = (_Float16)aReg[r][i];
      *(v4h*)(a + (arow + r * 32) * LDA + acol) = h;
    }
    _Float16* w = sW + buf * BN * LDW + nn * LDW + j2 * 8;
    const _Float16* lrow = sLut + nn * 16;
    #pragma unroll
    for (int u = 0; u < 2; ++u) {
      const uint32_t q = qReg[u];
      v8h h;
      #pragma unroll
      for (int b = 0; b < 8; ++b) h[b] = lrow[(q >> (4 * b)) & 15];
      *(v8h*)(w + u * 8) = h;            // 16 f16 contiguous in K per thread
    }
  };

  // A frag 16x32: lane = (Khalf<<4)|Mrow; two 16B chunks at K = half*8 and 16+half*8
  auto load_fragA = [&](const _Float16* a, int mb) -> v16h {
    const _Float16* p = a + (mb + (lane & 15)) * LDA + (lane >> 4) * 8;
    return make_v16(*(const v8h*)p, *(const v8h*)(p + 16));
  };
  // B frag 32x16 (=W^T): lane = (Khalf<<4)|Ncol; two 16B chunks at K = half*8 and 16+half*8
  auto load_fragB = [&](const _Float16* w, int nb) -> v16h {
    const _Float16* p = w + (nb + (lane & 15)) * LDW + (lane >> 4) * 8;
    return make_v16(*(const v8h*)p, *(const v8h*)(p + 16));
  };

  v8f acc[4][4];
  #pragma unroll
  for (int i = 0; i < 4; ++i)
    #pragma unroll
    for (int j = 0; j < 4; ++j)
      #pragma unroll
      for (int e = 0; e < 8; ++e) acc[i][j][e] = 0.0f;

  load_stage(0);
  __syncthreads();                        // sLut visible before first dequant
  write_stage(0);
  __syncthreads();                        // first tiles visible

  for (int ks = 0; ks < NKSTEP; ++ks) {
    const int buf = ks & 1;
    if (ks + 1 < NKSTEP) load_stage(ks + 1);   // global loads overlap WMMA

    const _Float16* a = sA + buf * BM * LDA;
    const _Float16* w = sW + buf * BN * LDW;
    v16h bf[4];
    #pragma unroll
    for (int nf = 0; nf < 4; ++nf) bf[nf] = load_fragB(w, wn + nf * 16);
    #pragma unroll
    for (int mf = 0; mf < 4; ++mf) {
      v16h af = load_fragA(a, wm + mf * 16);
      #pragma unroll
      for (int nf = 0; nf < 4; ++nf)
        acc[mf][nf] = __builtin_amdgcn_wmma_f32_16x16x32_f16(
            false, af, false, bf[nf], (short)0, acc[mf][nf], false, false);
    }

    if (ks + 1 < NKSTEP) write_stage(buf ^ 1); // fill other buffer (dequant + cvt)
    __syncthreads();
  }

  // ---- Epilogue: C/D 16x16 f32 layout -> global (lane N-contiguous, coalesced) ----
  const int cn   = n0 + wn + (lane & 15);
  const int mtop = m0 + wm + ((lane >> 4) << 3);
  #pragma unroll
  for (int mf = 0; mf < 4; ++mf) {
    #pragma unroll
    for (int nf = 0; nf < 4; ++nf) {
      float* cp = C + (size_t)(mtop + mf * 16) * N_ + cn + nf * 16;
      #pragma unroll
      for (int r = 0; r < 8; ++r)
        cp[(size_t)r * N_] = acc[mf][nf][r];
    }
  }
}

extern "C" void kernel_launch(void* const* d_in, const int* in_sizes, int n_in,
                              void* d_out, int out_size, void* d_ws, size_t ws_size,
                              hipStream_t stream) {
  const float* A = (const float*)d_in[0];   // [B,S,K] f32 -> [M,K]
  const int*   Q = (const int*)d_in[1];     // [K/8, N] int32
  const float* L = (const float*)d_in[2];   // [N, 16] f32
  float* C = (float*)d_out;                 // [M, N] f32
  (void)in_sizes; (void)n_in; (void)out_size; (void)d_ws; (void)ws_size;
  dim3 grid(N_ / BN, M_ / BM);              // (32, 32)
  sqllm_gemm_f16wmma<<<grid, dim3(256), SMEM_BYTES, stream>>>(A, Q, L, C);
}
